// EnhancedTCN_GNN_5351529250826
// MI455X (gfx1250) — matrix operand order repaired and checked by reading, not compiled, not run on hardware
//
#include <hip/hip_runtime.h>
#include <cstdint>

typedef _Float16 half_t;
typedef __attribute__((ext_vector_type(16))) _Float16 v16h;
typedef __attribute__((ext_vector_type(8)))  _Float16 v8h;
typedef __attribute__((ext_vector_type(8)))  float    v8f;

constexpr int cB   = 4;
constexpr int cT   = 24;
constexpr int cN   = 128;    // stations
constexpr int cFIN = 64;
constexpr int cC   = 256;
constexpr int cL   = 3;
constexpr int cK   = 3;
constexpr int cGH  = 2;
constexpr int cNH  = 4;
constexpr int cE   = 1024;
constexpr int cNN  = cB * cT * cN;        // 12288 nodes
constexpr int cBT  = cB * cT;             // 96 graphs
constexpr int cETOT = cBT * cE + cNN;     // 110592 edges incl. self loops

// ---------------------------------------------------------------------------
// WMMA fragment helpers (16x16x32 f16 layouts used in this file, wave32)
// A (16x32 f16): lane row M=lane&15; elem j: K = j + 16*(lane>=16)
//   -> two contiguous 8-half (128-bit) loads per fragment.
// B (32x16 f16): lane col N=lane&15; elem j: K = j + 16*(lane>=16)
//   -> one contiguous 16-half (two 128-bit) load per fragment.
// C/D (16x16 f32): elem r: M = r + 8*(lane>=16), N = lane&15.
// ---------------------------------------------------------------------------
__device__ inline v16h load_afrag(const half_t* __restrict__ row, int k0, int hi) {
  const v8h* p0 = (const v8h*)(row + k0 + hi * 16);
  const v8h* p1 = (const v8h*)(row + k0 + hi * 16 + 8);
  v8h lo = *p0;
  v8h hh = *p1;
  v16h a;
#pragma unroll
  for (int j = 0; j < 8; ++j) { a[j] = lo[j]; a[j + 8] = hh[j]; }
  return a;
}

// ---------------------------------------------------------------------------
// WMMA GEMM: out[M,N] = A[M,K] (f16) * Wt[N,K]^T (f16) + bias
// One wave per 32x64 output block: 2 M-tiles x 4 N-tiles = 8 WMMA per K-step.
// All 6 fragment loads (2 A + 4 B, 12 x b128) are issued into distinct
// registers before the WMMA burst so the whole step is one load clause
// followed by 8 chained WMMAs. flags: bit0 = store f16, bit1 = relu.
// Requires M % 32 == 0, N % 64 == 0, K % 32 == 0.
// ---------------------------------------------------------------------------
__global__ void k_gemm(const half_t* __restrict__ A, const half_t* __restrict__ Wt,
                       const float* __restrict__ bias, void* __restrict__ out,
                       int M, int K, int N, int flags) {
  const int lane = threadIdx.x;
  const int col  = lane & 15;
  const int hi   = lane >> 4;
  const int n0   = blockIdx.x * 64;
  const int m0   = blockIdx.y * 32;
  const half_t* arow0 = A  + (size_t)(m0 + col) * K;
  const half_t* arow1 = A  + (size_t)(m0 + 16 + col) * K;
  const half_t* brow  = Wt + (size_t)(n0 + col) * K + hi * 16;
  const size_t bstep = (size_t)16 * K;
  v8f acc[8];
  v8f zero = {};
#pragma unroll
  for (int i = 0; i < 8; ++i) acc[i] = zero;
  for (int k0 = 0; k0 < K; k0 += 32) {
    v16h a0 = load_afrag(arow0, k0, hi);
    v16h a1 = load_afrag(arow1, k0, hi);
    v16h b[4];
#pragma unroll
    for (int j = 0; j < 4; ++j) b[j] = *(const v16h*)(brow + (size_t)j * bstep + k0);
#pragma unroll
    for (int j = 0; j < 4; ++j)
      acc[j] = __builtin_amdgcn_wmma_f32_16x16x32_f16(false, a0, false, b[j], (short)0,
                                                      acc[j], false, false);
#pragma unroll
    for (int j = 0; j < 4; ++j)
      acc[j + 4] = __builtin_amdgcn_wmma_f32_16x16x32_f16(false, a1, false, b[j], (short)0,
                                                          acc[j + 4], false, false);
  }
#pragma unroll
  for (int j = 0; j < 4; ++j) {
    const int n = n0 + j * 16 + col;
    const float bv = bias ? bias[n] : 0.0f;
#pragma unroll
    for (int mi = 0; mi < 2; ++mi) {
      v8f acv = acc[j + 4 * mi];
#pragma unroll
      for (int r = 0; r < 8; ++r) {
        const int m = m0 + mi * 16 + r + hi * 8;
        float v = acv[r] + bv;
        if (flags & 2) v = v > 0.0f ? v : 0.0f;
        if (flags & 1) ((half_t*)out)[(size_t)m * N + n] = (half_t)v;
        else           ((float*)out)[(size_t)m * N + n] = v;
      }
    }
  }
}

// ---------------------------------------------------------------------------
// Attention: one wave per (bt, head, 16-row q tile). Scores = 8 WMMA tiles
// (K=64), in-register softmax, P->LDS f16, P@V = 4 WMMA tiles (K=128).
// ---------------------------------------------------------------------------
__global__ void k_attn(const half_t* __restrict__ q16, const half_t* __restrict__ k16,
                       const half_t* __restrict__ v16t, half_t* __restrict__ attn16) {
  __shared__ __align__(32) half_t Pl[16 * 128];
  const int idx = blockIdx.x;
  const int qt = idx & 7;
  const int h  = (idx >> 3) & 3;
  const int bt = idx >> 5;
  const int lane = threadIdx.x;
  const int col  = lane & 15;
  const int hi   = lane >> 4;
  v8f zero = {};
  v8f acc[8];
#pragma unroll
  for (int i = 0; i < 8; ++i) acc[i] = zero;
  const half_t* qrow = q16 + (size_t)(bt * cN + qt * 16 + col) * cC + h * 64;
  const half_t* kbase = k16 + (size_t)(bt * cN + col) * cC + h * 64 + hi * 16;
  for (int k0 = 0; k0 < 64; k0 += 32) {
    v16h a = load_afrag(qrow, k0, hi);
    v16h b[4];
#pragma unroll
    for (int half4 = 0; half4 < 2; ++half4) {
#pragma unroll
      for (int j = 0; j < 4; ++j)
        b[j] = *(const v16h*)(kbase + (size_t)(half4 * 4 + j) * 16 * cC + k0);
#pragma unroll
      for (int j = 0; j < 4; ++j)
        acc[half4 * 4 + j] = __builtin_amdgcn_wmma_f32_16x16x32_f16(
            false, a, false, b[j], (short)0, acc[half4 * 4 + j], false, false);
    }
  }
  const float scale = 0.125f;  // 1/sqrt(64)
  float rowsum[8];
#pragma unroll
  for (int r = 0; r < 8; ++r) {
    float mx = -1e30f;
#pragma unroll
    for (int nt = 0; nt < 8; ++nt) mx = fmaxf(mx, acc[nt][r] * scale);
    for (int o = 8; o >= 1; o >>= 1) mx = fmaxf(mx, __shfl_xor(mx, o));
    float sm = 0.0f;
#pragma unroll
    for (int nt = 0; nt < 8; ++nt) {
      float p = __expf(acc[nt][r] * scale - mx);
      sm += p;
      Pl[(r + 8 * hi) * 128 + nt * 16 + col] = (half_t)p;
    }
    for (int o = 8; o >= 1; o >>= 1) sm += __shfl_xor(sm, o);
    rowsum[r] = sm;
  }
  __syncthreads();  // single-wave: ordering fence for LDS write->read
  const half_t* prow = &Pl[col * 128];
#pragma unroll
  for (int dt = 0; dt < 4; ++dt) {
    v8f ov = zero;
    const half_t* vrow = v16t + (size_t)((bt * cNH + h) * 64 + dt * 16 + col) * 128 + hi * 16;
    for (int k0 = 0; k0 < 128; k0 += 32) {
      v16h a = load_afrag(prow, k0, hi);
      v16h b = *(const v16h*)(vrow + k0);
      ov = __builtin_amdgcn_wmma_f32_16x16x32_f16(false, a, false, b, (short)0, ov,
                                                  false, false);
    }
#pragma unroll
    for (int r = 0; r < 8; ++r) {
      float val = ov[r] / rowsum[r];
      attn16[(size_t)(bt * cN + qt * 16 + r + 8 * hi) * cC + h * 64 + dt * 16 + col] =
          (half_t)val;
    }
  }
}

// ---------------------------------------------------------------------------
// Small utility kernels
// ---------------------------------------------------------------------------
__global__ void k_cvt(const float* __restrict__ a, half_t* __restrict__ o, int n) {
  int i = blockIdx.x * blockDim.x + threadIdx.x;
  if (i < n) o[i] = (half_t)a[i];
}

__global__ void k_wt(const float* __restrict__ W, half_t* __restrict__ Wt, int K, int N) {
  int i = blockIdx.x * blockDim.x + threadIdx.x;
  if (i >= K * N) return;
  int k = i / N, n = i - k * N;
  Wt[(size_t)n * K + k] = (half_t)W[i];
}

__global__ void k_fill(float* __restrict__ p, float v, int n) {
  int i = blockIdx.x * blockDim.x + threadIdx.x;
  if (i < n) p[i] = v;
}

__global__ void k_enc_post(const float* __restrict__ henc, const float* __restrict__ st,
                           const float* __restrict__ hz, float* __restrict__ hbuf,
                           half_t* __restrict__ h16, int total) {
  int i = blockIdx.x * blockDim.x + threadIdx.x;
  if (i >= total) return;
  int c = i & (cC - 1);
  int node = i >> 8;
  int n = node & (cN - 1);
  int t = (node >> 7) % cT;
  float v = henc[i] + st[n * cC + c] + hz[t * cC + c];
  hbuf[i] = v;
  h16[i] = (half_t)v;
}

__global__ void k_im2col(const float* __restrict__ hbuf, half_t* __restrict__ A,
                         int dil, int total) {
  int i = blockIdx.x * blockDim.x + threadIdx.x;
  if (i >= total) return;
  int col = i % (cC * cK);
  int row = i / (cC * cK);
  int ci = col / cK;
  int k  = col - ci * cK;
  int t  = (row >> 7) % cT;
  int tt = t - (cK - 1 - k) * dil;
  float v = 0.0f;
  if (tt >= 0) v = hbuf[(size_t)(row + (tt - t) * cN) * cC + ci];
  A[i] = (half_t)v;
}

__global__ void k_conv_post(const float* __restrict__ ybuf, float* __restrict__ skips,
                            float* __restrict__ hbuf, half_t* __restrict__ h16, int total) {
  int i = blockIdx.x * blockDim.x + threadIdx.x;
  if (i >= total) return;
  float y = ybuf[i];            // already relu(conv + bias)
  skips[i] += y;
  float v = hbuf[i] + y;
  hbuf[i] = v;
  h16[i] = (half_t)v;
}

__device__ inline void edge_nodes(const int* __restrict__ edge, int e, int& src, int& dst) {
  if (e < cBT * cE) {
    int g = e / cE, ee = e - g * cE;
    src = g * cN + edge[ee];
    dst = g * cN + edge[cE + ee];
  } else {
    src = dst = e - cBT * cE;
  }
}

__global__ void k_gat_logits(const float* __restrict__ xl, const float* __restrict__ xr,
                             const float* __restrict__ att, const int* __restrict__ edge,
                             float* __restrict__ logit, float* __restrict__ mseg) {
  const int idx = blockIdx.x;
  const int h = idx & 1, e = idx >> 1;
  const int lane = threadIdx.x;
  int src, dst;
  edge_nodes(edge, e, src, dst);
  const float* pl = xl + (size_t)src * (cGH * cC) + h * cC;
  const float* pr = xr + (size_t)dst * (cGH * cC) + h * cC;
  const float* pa = att + h * cC;
  float s = 0.0f;
#pragma unroll
  for (int i = 0; i < 8; ++i) {
    int c = lane + i * 32;
    float v = pl[c] + pr[c];
    v = v > 0.0f ? v : 0.2f * v;   // leaky relu
    s += v * pa[c];
  }
  for (int o = 16; o >= 1; o >>= 1) s += __shfl_xor(s, o);
  if (lane == 0) {
    logit[idx] = s;
    unsigned* addr = (unsigned*)&mseg[dst * cGH + h];
    unsigned old = *addr;
    while (__uint_as_float(old) < s) {
      unsigned assumed = old;
      old = atomicCAS(addr, assumed, __float_as_uint(s));
      if (old == assumed) break;
    }
  }
}

__global__ void k_gat_agg(const float* __restrict__ xl, const int* __restrict__ edge,
                          const float* __restrict__ logit, const float* __restrict__ mseg,
                          float* __restrict__ den, float* __restrict__ agg) {
  const int idx = blockIdx.x;
  const int h = idx & 1, e = idx >> 1;
  const int lane = threadIdx.x;
  int src, dst;
  edge_nodes(edge, e, src, dst);
  float w = __expf(logit[idx] - mseg[dst * cGH + h]);
  if (lane == 0) atomicAdd(&den[dst * cGH + h], w);
  const float* pl = xl + (size_t)src * (cGH * cC) + h * cC;
  float* pa = agg + (size_t)(dst * cGH + h) * cC;
#pragma unroll
  for (int i = 0; i < 8; ++i) {
    int c = lane + i * 32;
    atomicAdd(&pa[c], pl[c] * w);
  }
}

__global__ void k_gat_fin(const float* __restrict__ agg, const float* __restrict__ den,
                          const float* __restrict__ gatb, const float* __restrict__ ng,
                          const float* __restrict__ nb, float* __restrict__ hbuf,
                          half_t* __restrict__ h16) {
  const int n = blockIdx.x;
  const int lane = threadIdx.x;
  const float d0 = den[n * 2 + 0] + 1e-16f;
  const float d1 = den[n * 2 + 1] + 1e-16f;
  float t[8];
  float s = 0.0f, ss = 0.0f;
#pragma unroll
  for (int i = 0; i < 8; ++i) {
    int c = lane + i * 32;
    float g = 0.5f * (agg[(size_t)(n * 2) * cC + c] / d0 +
                      agg[(size_t)(n * 2 + 1) * cC + c] / d1) + gatb[c];
    float v = hbuf[(size_t)n * cC + c] + g;
    t[i] = v;
    s += v;
    ss += v * v;
  }
  for (int o = 16; o >= 1; o >>= 1) { s += __shfl_xor(s, o); ss += __shfl_xor(ss, o); }
  float m  = s * (1.0f / cC);
  float var = ss * (1.0f / cC) - m * m;
  float rs = rsqrtf(var + 1e-5f);
#pragma unroll
  for (int i = 0; i < 8; ++i) {
    int c = lane + i * 32;
    float o = (t[i] - m) * rs * ng[c] + nb[c];
    hbuf[(size_t)n * cC + c] = o;
    h16[(size_t)n * cC + c] = (half_t)o;
  }
}

__global__ void k_add_cvt(const float* __restrict__ addv, float* __restrict__ hbuf,
                          half_t* __restrict__ h16, int total) {
  int i = blockIdx.x * blockDim.x + threadIdx.x;
  if (i >= total) return;
  float v = hbuf[i] + addv[i];
  hbuf[i] = v;
  h16[i] = (half_t)v;
}

__global__ void k_vtrans(const float* __restrict__ vbuf, half_t* __restrict__ v16t,
                         int total) {
  int i = blockIdx.x * blockDim.x + threadIdx.x;
  if (i >= total) return;
  int c = i & (cC - 1);
  int node = i >> 8;
  int bt = node >> 7;
  int n = node & (cN - 1);
  int h = c >> 6;
  int d = c & 63;
  v16t[(size_t)((bt * cNH + h) * 64 + d) * 128 + n] = (half_t)vbuf[i];
}

__global__ void k_add_ln(const float* __restrict__ addv, const float* __restrict__ g,
                         const float* __restrict__ b, float* __restrict__ hbuf,
                         half_t* __restrict__ h16) {
  const int n = blockIdx.x;
  const int lane = threadIdx.x;
  float t[8];
  float s = 0.0f, ss = 0.0f;
#pragma unroll
  for (int i = 0; i < 8; ++i) {
    int c = lane + i * 32;
    float v = hbuf[(size_t)n * cC + c] + addv[(size_t)n * cC + c];
    t[i] = v;
    s += v;
    ss += v * v;
  }
  for (int o = 16; o >= 1; o >>= 1) { s += __shfl_xor(s, o); ss += __shfl_xor(ss, o); }
  float m  = s * (1.0f / cC);
  float var = ss * (1.0f / cC) - m * m;
  float rs = rsqrtf(var + 1e-5f);
#pragma unroll
  for (int i = 0; i < 8; ++i) {
    int c = lane + i * 32;
    float o = (t[i] - m) * rs * g[c] + b[c];
    hbuf[(size_t)n * cC + c] = o;
    h16[(size_t)n * cC + c] = (half_t)o;
  }
}

__global__ void k_final(const float* __restrict__ href, const float* __restrict__ gbuf,
                        const float* __restrict__ sbuf, const float* __restrict__ wout,
                        const float* __restrict__ bout, float* __restrict__ out) {
  const int n = blockIdx.x;
  const int lane = threadIdx.x;
  float a0 = 0.0f, a1 = 0.0f;
#pragma unroll
  for (int i = 0; i < 8; ++i) {
    int c = lane + i * 32;
    float gl = gbuf[(size_t)n * cC + c];
    float g = 1.0f / (1.0f + __expf(-gl));
    float hf = g * href[(size_t)n * cC + c] + (1.0f - g) * sbuf[(size_t)n * cC + c];
    a0 += hf * wout[c * 2 + 0];
    a1 += hf * wout[c * 2 + 1];
  }
  for (int o = 16; o >= 1; o >>= 1) { a0 += __shfl_xor(a0, o); a1 += __shfl_xor(a1, o); }
  if (lane == 0) {
    out[n * 2 + 0] = a0 + bout[0];
    float z = a1 + bout[1];
    out[n * 2 + 1] = (z > 20.0f) ? z : log1pf(__expf(z));
  }
}

// ---------------------------------------------------------------------------
extern "C" void kernel_launch(void* const* d_in, const int* in_sizes, int n_in,
                              void* d_out, int out_size, void* d_ws, size_t ws_size,
                              hipStream_t stream) {
  (void)in_sizes; (void)n_in; (void)out_size; (void)ws_size;
  const float* x       = (const float*)d_in[0];
  const int*   edge    = (const int*)  d_in[1];
  const float* W_enc   = (const float*)d_in[2];
  const float* b_enc   = (const float*)d_in[3];
  const float* st_emb  = (const float*)d_in[4];
  const float* hz_emb  = (const float*)d_in[5];
  const float* conv_w  = (const float*)d_in[6];
  const float* conv_b  = (const float*)d_in[7];
  const float* gat_wl  = (const float*)d_in[8];
  const float* gat_wr  = (const float*)d_in[9];
  const float* gat_att = (const float*)d_in[10];
  const float* gat_b   = (const float*)d_in[11];
  const float* norm_g  = (const float*)d_in[12];
  const float* norm_b  = (const float*)d_in[13];
  const float* Wq = (const float*)d_in[14];
  const float* Wk = (const float*)d_in[15];
  const float* Wv = (const float*)d_in[16];
  const float* Wo = (const float*)d_in[17];
  const float* bq = (const float*)d_in[18];
  const float* bk = (const float*)d_in[19];
  const float* bv = (const float*)d_in[20];
  const float* bo = (const float*)d_in[21];
  const float* an_g = (const float*)d_in[22];
  const float* an_b = (const float*)d_in[23];
  const float* W_skip = (const float*)d_in[24];
  const float* b_skip = (const float*)d_in[25];
  const float* W_gate = (const float*)d_in[26];
  const float* b_gate = (const float*)d_in[27];
  const float* W_out  = (const float*)d_in[28];
  const float* b_out  = (const float*)d_in[29];
  float* out = (float*)d_out;

  // ---- workspace bump allocator (with aliasing)
  char* wsp = (char*)d_ws;
  size_t off = 0;
  auto take = [&](size_t bytes) -> char* {
    char* p = wsp + off;
    off += (bytes + 255) & ~(size_t)255;
    return p;
  };
  half_t* wenc_t  = (half_t*)take((size_t)cC * cFIN * 2);
  half_t* wconv_t = (half_t*)take((size_t)cL * cC * (cC * cK) * 2);
  half_t* wl_t    = (half_t*)take((size_t)cL * (cGH * cC) * cC * 2);
  half_t* wr_t    = (half_t*)take((size_t)cL * (cGH * cC) * cC * 2);
  half_t* wq_t    = (half_t*)take((size_t)cC * cC * 2);
  half_t* wk_t    = (half_t*)take((size_t)cC * cC * 2);
  half_t* wv_t    = (half_t*)take((size_t)cC * cC * 2);
  half_t* wo_t    = (half_t*)take((size_t)cC * cC * 2);
  half_t* wg_t    = (half_t*)take((size_t)cC * cC * 2);
  half_t* wsk_t   = (half_t*)take((size_t)cC * cFIN * 2);
  half_t* x16     = (half_t*)take((size_t)cNN * cFIN * 2);
  float*  hbuf    = (float*) take((size_t)cNN * cC * 4);   // h_tcn -> hd -> h_ref
  half_t* h16     = (half_t*)take((size_t)cNN * cC * 2);   // f16 mirror of hbuf
  float*  bufA    = (float*) take((size_t)cNN * cC * 4);   // skipsum | ao
  float*  bufB    = (float*) take((size_t)cNN * cC * 4);   // ybuf | vbuf | gbuf
  half_t* bufC    = (half_t*)take((size_t)cNN * (cC * cK) * 2); // im2col | q16,k16,v16t
  float*  bufD    = (float*) take((size_t)cNN * (cGH * cC) * 4); // xl | sbuf+attn16
  float*  bufE    = (float*) take((size_t)cNN * (cGH * cC) * 4); // xr | agg
  float*  logit   = (float*) take((size_t)cETOT * cGH * 4);
  float*  mseg    = (float*) take((size_t)cNN * cGH * 4);
  float*  den     = (float*) take((size_t)cNN * cGH * 4);

  float*  skips  = bufA;
  float*  aobuf  = bufA;
  float*  ybuf   = bufB;
  float*  vbuf   = bufB;
  float*  gbuf   = bufB;
  half_t* A768   = bufC;
  half_t* q16    = bufC;
  half_t* k16    = bufC + (size_t)cNN * cC;
  half_t* v16t   = bufC + (size_t)2 * cNN * cC;
  float*  xl     = bufD;
  float*  sbuf   = bufD;
  half_t* attn16 = (half_t*)(bufD + (size_t)cNN * cC);
  float*  xr     = bufE;
  float*  agg    = bufE;

  const int TB = 256;
  auto g1 = [&](int n) { return dim3((unsigned)((n + TB - 1) / TB)); };
  dim3 gemmC(cC / 64, cNN / 32);          // N=256 GEMMs
  dim3 gemmG((cGH * cC) / 64, cNN / 32);  // N=512 GEMMs

  // ---- weight prep (f32 -> transposed f16 [N,K])
  k_wt<<<g1(cFIN * cC), TB, 0, stream>>>(W_enc, wenc_t, cFIN, cC);
  k_cvt<<<g1(cL * cC * cC * cK), TB, 0, stream>>>(conv_w, wconv_t, cL * cC * cC * cK);
  for (int l = 0; l < cL; ++l) {
    k_wt<<<g1(cC * cGH * cC), TB, 0, stream>>>(gat_wl + (size_t)l * cC * cGH * cC,
                                               wl_t + (size_t)l * cGH * cC * cC, cC, cGH * cC);
    k_wt<<<g1(cC * cGH * cC), TB, 0, stream>>>(gat_wr + (size_t)l * cC * cGH * cC,
                                               wr_t + (size_t)l * cGH * cC * cC, cC, cGH * cC);
  }
  k_wt<<<g1(cC * cC), TB, 0, stream>>>(Wq, wq_t, cC, cC);
  k_wt<<<g1(cC * cC), TB, 0, stream>>>(Wk, wk_t, cC, cC);
  k_wt<<<g1(cC * cC), TB, 0, stream>>>(Wv, wv_t, cC, cC);
  k_wt<<<g1(cC * cC), TB, 0, stream>>>(Wo, wo_t, cC, cC);
  k_wt<<<g1(cC * cC), TB, 0, stream>>>(W_gate, wg_t, cC, cC);
  k_wt<<<g1(cFIN * cC), TB, 0, stream>>>(W_skip, wsk_t, cFIN, cC);
  k_cvt<<<g1(cNN * cFIN), TB, 0, stream>>>(x, x16, cNN * cFIN);

  // ---- encoder + embeddings
  k_gemm<<<gemmC, 32, 0, stream>>>(x16, wenc_t, b_enc, ybuf, cNN, cFIN, cC, 0);
  k_enc_post<<<g1(cNN * cC), TB, 0, stream>>>(ybuf, st_emb, hz_emb, hbuf, h16, cNN * cC);
  k_fill<<<g1(cNN * cC), TB, 0, stream>>>(skips, 0.0f, cNN * cC);

  // ---- TCN + GAT layers
  for (int l = 0; l < cL; ++l) {
    const int dil = 1 << l;
    k_im2col<<<g1(cNN * cC * cK), TB, 0, stream>>>(hbuf, A768, dil, cNN * cC * cK);
    k_gemm<<<gemmC, 32, 0, stream>>>(A768, wconv_t + (size_t)l * cC * cC * cK,
                                     conv_b + l * cC, ybuf, cNN, cC * cK, cC, 2 /*relu*/);
    k_conv_post<<<g1(cNN * cC), TB, 0, stream>>>(ybuf, skips, hbuf, h16, cNN * cC);
    k_gemm<<<gemmG, 32, 0, stream>>>(h16, wl_t + (size_t)l * cGH * cC * cC, nullptr, xl,
                                     cNN, cC, cGH * cC, 0);
    k_gemm<<<gemmG, 32, 0, stream>>>(h16, wr_t + (size_t)l * cGH * cC * cC, nullptr, xr,
                                     cNN, cC, cGH * cC, 0);
    k_fill<<<g1(cNN * cGH), TB, 0, stream>>>(mseg, -1e30f, cNN * cGH);
    k_fill<<<g1(cNN * cGH), TB, 0, stream>>>(den, 0.0f, cNN * cGH);
    k_gat_logits<<<cETOT * cGH, 32, 0, stream>>>(xl, xr, gat_att + (size_t)l * cGH * cC,
                                                 edge, logit, mseg);
    k_fill<<<g1(cNN * cGH * cC), TB, 0, stream>>>(agg, 0.0f, cNN * cGH * cC);  // aliases xr
    k_gat_agg<<<cETOT * cGH, 32, 0, stream>>>(xl, edge, logit, mseg, den, agg);
    k_gat_fin<<<cNN, 32, 0, stream>>>(agg, den, gat_b + l * cC, norm_g + l * cC,
                                      norm_b + l * cC, hbuf, h16);
  }

  // ---- h_deep = skipsum + h_tcn
  k_add_cvt<<<g1(cNN * cC), TB, 0, stream>>>(skips, hbuf, h16, cNN * cC);

  // ---- MHA
  k_gemm<<<gemmC, 32, 0, stream>>>(h16, wq_t, bq, q16, cNN, cC, cC, 1 /*f16 out*/);
  k_gemm<<<gemmC, 32, 0, stream>>>(h16, wk_t, bk, k16, cNN, cC, cC, 1);
  k_gemm<<<gemmC, 32, 0, stream>>>(h16, wv_t, bv, vbuf, cNN, cC, cC, 0);
  k_vtrans<<<g1(cNN * cC), TB, 0, stream>>>(vbuf, v16t, cNN * cC);
  k_attn<<<cBT * cNH * 8, 32, 0, stream>>>(q16, k16, v16t, attn16);
  k_gemm<<<gemmC, 32, 0, stream>>>(attn16, wo_t, bo, aobuf, cNN, cC, cC, 0);
  k_add_ln<<<cNN, 32, 0, stream>>>(aobuf, an_g, an_b, hbuf, h16);

  // ---- gating + output head
  k_gemm<<<gemmC, 32, 0, stream>>>(h16, wg_t, b_gate, gbuf, cNN, cC, cC, 0);
  k_gemm<<<gemmC, 32, 0, stream>>>(x16, wsk_t, b_skip, sbuf, cNN, cFIN, cC, 0);
  k_final<<<cNN, 32, 0, stream>>>(hbuf, gbuf, sbuf, W_out, b_out, out);
}
